// CustomBlock_71159018160683
// MI455X (gfx1250) — matrix operand order repaired and checked by reading, not compiled, not run on hardware
//
#include <hip/hip_runtime.h>
#include <math.h>

// ---------------------------------------------------------------------------
// CDNA5 (gfx1250) dual attention block (self + cross).
// - All GEMMs via v_wmma_f32_16x16x32_bf16 (wave32), ping/pong operand regs
//   (compile-time selection -> no scratch spills).
// - TDM (tensor_load_to_lds + s_wait_tensorcnt) stages A/Q panels into LDS.
// - Attention keeps full 16x2048 f32 score panel in LDS for exact softmax.
// ---------------------------------------------------------------------------

typedef __bf16 bf16;
typedef __attribute__((ext_vector_type(16))) __bf16 v16bf;
typedef __attribute__((ext_vector_type(8)))  __bf16 v8bf;
typedef __attribute__((ext_vector_type(8)))  float  v8f;
typedef __attribute__((ext_vector_type(4)))  unsigned int u32x4;
typedef __attribute__((ext_vector_type(8)))  int i32x8;
typedef __attribute__((ext_vector_type(4)))  int i32x4;

#define BATCH 8
#define NSEQ  2048
#define DA    768
#define DB    1024
#define MROWS (BATCH * NSEQ)   // 16384

union ABPack { v16bf v; v8bf h[2]; };

__device__ __forceinline__ v8f wmma_bf16(v16bf a, v16bf b, v8f c) {
  // (neg_a, A, neg_b, B, c_mod, C, reuse_a, reuse_b)
  return __builtin_amdgcn_wmma_f32_16x16x32_bf16(false, a, false, b,
                                                 (short)0, c, false, false);
}

// ---------------------------------------------------------------------------
// TDM: DMA a 2D bf16 tile [tile_d1 rows x tile_d0 cols] from global to LDS.
// D# packing per CDNA5 ISA ch.8 (group0: count/lds_addr/global_addr/type=2;
// group1: data_size=2B, tensor dims, tile dims, dim0 stride).
// ---------------------------------------------------------------------------
__device__ __forceinline__ void tdm_load_2d_bf16(
    void* lds_dst, const void* gsrc,
    unsigned tile_d0, unsigned tile_d1,
    unsigned tensor_d0, unsigned tensor_d1, unsigned stride_d0) {
  unsigned long long ga = (unsigned long long)(uintptr_t)gsrc;
  u32x4 g0;
  g0[0] = 1u;                                              // count=1 (user mode)
  g0[1] = (unsigned)(uintptr_t)lds_dst;                    // LDS byte address
  g0[2] = (unsigned)(ga & 0xFFFFFFFFu);                    // global_addr[31:0]
  g0[3] = (unsigned)((ga >> 32) & 0x1FFFFFFu) | (2u << 30);// addr[56:32], type=2
  i32x8 g1;
  g1[0] = (int)(1u << 16);                                 // data_size=1 (2 bytes)
  g1[1] = (int)((tensor_d0 & 0xFFFFu) << 16);              // tensor_dim0[15:0]
  g1[2] = (int)((tensor_d0 >> 16) | ((tensor_d1 & 0xFFFFu) << 16));
  g1[3] = (int)((tensor_d1 >> 16) | (tile_d0 << 16));      // tile_dim0
  g1[4] = (int)tile_d1;                                    // tile_dim1, tile_dim2=0
  g1[5] = (int)stride_d0;                                  // dim0 stride[31:0]
  g1[6] = 0;
  g1[7] = 0;
  i32x4 z = {0, 0, 0, 0};
#if defined(__clang_major__) && __clang_major__ >= 23
  i32x8 z8 = {0, 0, 0, 0, 0, 0, 0, 0};
  __builtin_amdgcn_tensor_load_to_lds(g0, g1, z, z, z8, 0);
#else
  __builtin_amdgcn_tensor_load_to_lds(g0, g1, z, z, 0);
#endif
}

// ---------------------------------------------------------------------------
// f32 -> bf16 elementwise convert
// ---------------------------------------------------------------------------
__global__ __launch_bounds__(256) void cvt_f32_bf16_kernel(
    const float* __restrict__ x, bf16* __restrict__ y, int n) {
  int i = blockIdx.x * 256 + threadIdx.x;
  if (i < n) y[i] = (bf16)x[i];
}

// ---------------------------------------------------------------------------
// W[K][N] f32 -> Wt[N][K] bf16  (so GEMM B-operand columns are contiguous)
// ---------------------------------------------------------------------------
__global__ __launch_bounds__(256) void transpose_cvt_kernel(
    const float* __restrict__ W, bf16* __restrict__ Wt, int K, int Nc) {
  int n = blockIdx.x * 16 + (threadIdx.x & 15);
  int k = blockIdx.y * 16 + (threadIdx.x >> 4);
  if (n < Nc && k < K) Wt[(size_t)n * K + k] = (bf16)W[(size_t)k * Nc + n];
}

// ---------------------------------------------------------------------------
// GEMM: out = X[M,K](bf16) @ W + bias, W pre-transposed Wt[Nc,K] bf16.
// WG: 256 thr = 8 waves as 2(M) x 4(N); wave tile 16x64; WG tile 32x256.
// A panel (32 x K bf16, <=64KB) staged in LDS via TDM, reused by 4 N-waves.
// Ping/pong named operand registers; K is a multiple of 64.
// ---------------------------------------------------------------------------
__global__ __launch_bounds__(256) void gemm_bias_kernel(
    const bf16* __restrict__ X, const bf16* __restrict__ Wt,
    const float* __restrict__ bias,
    bf16* __restrict__ outb, float* __restrict__ outf,
    const float* __restrict__ resid, bf16* __restrict__ outT,
    int K, int Nc, int Nseq) {
  extern __shared__ char smem_g[];
  bf16* As = (bf16*)smem_g;                    // [32][K]

  const int lane = threadIdx.x & 31;
  const int wave = threadIdx.x >> 5;
  const int wm = wave & 1, wn = wave >> 1;
  const int m0 = blockIdx.x * 32 + wm * 16;
  const int n0 = blockIdx.y * 256 + wn * 64;
  const int half = lane >> 4;

  if (wave == 0) {
    tdm_load_2d_bf16(As, X + (size_t)blockIdx.x * 32 * K,
                     (unsigned)K, 32u, (unsigned)K, (unsigned)MROWS, (unsigned)K);
    __builtin_amdgcn_s_wait_tensorcnt(0);
  }
  __syncthreads();

  const bf16* arow = As + (size_t)(wm * 16 + (lane & 15)) * K;

  auto loadA = [&](int k, ABPack& aa) {
    const bf16* ap = arow + k + half * 8;      // two 8-elt runs (A layout)
    aa.h[0] = *(const v8bf*)ap;
    aa.h[1] = *(const v8bf*)(ap + 16);
  };
  auto loadB = [&](int k, ABPack* bb) {
#pragma unroll
    for (int t = 0; t < 4; ++t) {
      const int col = n0 + t * 16 + (lane & 15);
      const bf16* bp = Wt + (size_t)col * K + k + half * 8;
      bb[t].h[0] = *(const v8bf*)bp;
      bb[t].h[1] = *(const v8bf*)(bp + 16);
      __builtin_prefetch(bp + 128, 0, 1);
    }
  };

  v8f acc[4] = {};
  ABPack a0, a1;
  ABPack b0[4], b1[4];
  loadA(0, a0);
  loadB(0, b0);
  for (int k = 0; k < K; k += 64) {            // K % 64 == 0 (1024, 768)
    loadA(k + 32, a1);
    loadB(k + 32, b1);
#pragma unroll
    for (int t = 0; t < 4; ++t) acc[t] = wmma_bf16(a0.v, b0[t].v, acc[t]);
    if (k + 64 < K) {
      loadA(k + 64, a0);
      loadB(k + 64, b0);
    }
#pragma unroll
    for (int t = 0; t < 4; ++t) acc[t] = wmma_bf16(a1.v, b1[t].v, acc[t]);
  }

  // Epilogue. C layout: lane<16 -> (M=r, N=lane); lane>=16 -> (M=r+8, N=lane-16)
#pragma unroll
  for (int t = 0; t < 4; ++t) {
    const int n = n0 + t * 16 + (lane & 15);
    const float bv = bias[n];
#pragma unroll
    for (int r = 0; r < 8; ++r) {
      const int m = m0 + r + half * 8;
      float v = acc[t][r] + bv;
      if (resid) v += resid[(size_t)m * Nc + n];
      if (outf) outf[(size_t)m * Nc + n] = v;
      if (outb) outb[(size_t)m * Nc + n] = (bf16)v;
      if (outT) {
        const int bb = m / Nseq, q = m % Nseq;
        outT[((size_t)bb * Nc + n) * Nseq + q] = (bf16)v;
      }
    }
  }
}

// ---------------------------------------------------------------------------
// Attention: O = softmax(Q K^T * inv_scale) V for one 16-row query tile.
// Q,K: [B*N, DB] bf16 row-major; Vt: [B][DB][N] bf16; O: [B*N, DB] bf16.
// LDS: Q panel 16x1024 bf16 (32KB, TDM-staged) + 16x2048 f32 scores (128KB).
// ---------------------------------------------------------------------------
__global__ __launch_bounds__(256) void attn_kernel(
    const bf16* __restrict__ Q, const bf16* __restrict__ Km,
    const bf16* __restrict__ Vt, bf16* __restrict__ O, float inv_scale) {
  extern __shared__ char smem_raw[];
  bf16*  Qs = (bf16*)smem_raw;                             // 32KB
  float* Ss = (float*)(smem_raw + 16 * DB * sizeof(bf16)); // 128KB

  const int tid = threadIdx.x;
  const int lane = tid & 31;
  const int wave = tid >> 5;
  const int half = lane >> 4;
  const int b = blockIdx.x >> 7;      // 128 query tiles per batch
  const int qrow0 = (blockIdx.x & 127) * 16;

  if (wave == 0) {
    tdm_load_2d_bf16(Qs, Q + ((size_t)(b * NSEQ + qrow0)) * DB,
                     (unsigned)DB, 16u, (unsigned)DB, (unsigned)MROWS,
                     (unsigned)DB);
    __builtin_amdgcn_s_wait_tensorcnt(0);
  }
  __syncthreads();

  // ---- Score phase: S[16, 2048] = Q K^T * inv_scale (ping/pong regs) ----
  for (int j = wave; j < NSEQ / 16; j += 8) {
    const int kcol = j * 16 + (lane & 15);  // key index = B-matrix column
    const bf16* qrow = Qs + (size_t)(lane & 15) * DB;
    const bf16* krow = Km + ((size_t)(b * NSEQ + kcol)) * DB;
    auto loadQA = [&](int d, ABPack& aa) {
      const bf16* ap = qrow + d + half * 8;
      aa.h[0] = *(const v8bf*)ap;
      aa.h[1] = *(const v8bf*)(ap + 16);
    };
    auto loadKB = [&](int d, ABPack& bb) {
      const bf16* bp = krow + d + half * 8;
      bb.h[0] = *(const v8bf*)bp;
      bb.h[1] = *(const v8bf*)(bp + 16);
    };
    v8f acc = {};
    ABPack a0, a1, b0, b1;
    loadQA(0, a0);
    loadKB(0, b0);
    for (int d = 0; d < DB; d += 64) {        // DB % 64 == 0
      loadQA(d + 32, a1);
      loadKB(d + 32, b1);
      acc = wmma_bf16(a0.v, b0.v, acc);
      if (d + 64 < DB) {
        loadQA(d + 64, a0);
        loadKB(d + 64, b0);
      }
      acc = wmma_bf16(a1.v, b1.v, acc);
    }
#pragma unroll
    for (int r = 0; r < 8; ++r)
      Ss[(r + half * 8) * NSEQ + j * 16 + (lane & 15)] = acc[r] * inv_scale;
  }
  __syncthreads();

  // ---- Exact softmax over each full row (16 threads per row) ----
  {
    const int rrow = tid >> 4, slot = tid & 15;
    float* srow = Ss + rrow * NSEQ;
    float mx = -3.402823466e38f;
    for (int c = slot; c < NSEQ; c += 16) mx = fmaxf(mx, srow[c]);
#pragma unroll
    for (int off = 8; off; off >>= 1) mx = fmaxf(mx, __shfl_xor(mx, off, 16));
    float sum = 0.f;
    for (int c = slot; c < NSEQ; c += 16) {
      float e = __expf(srow[c] - mx);
      srow[c] = e;
      sum += e;
    }
#pragma unroll
    for (int off = 8; off; off >>= 1) sum += __shfl_xor(sum, off, 16);
    const float inv = 1.f / sum;
    for (int c = slot; c < NSEQ; c += 16) srow[c] *= inv;
  }
  __syncthreads();

  // ---- PV phase: wave owns output cols [wave*128, wave*128+128) ----
  v8f oacc[8] = {};
  const float* prow = Ss + (size_t)(lane & 15) * NSEQ;
  for (int k0 = 0; k0 < NSEQ; k0 += 32) {
    v16bf a;  // P tile 16x32 from LDS (two 32B ds reads -> bf16 cvt)
    {
      const int base = half * 8;
      const v8f r0 = *(const v8f*)(prow + k0 + base);
      const v8f r1 = *(const v8f*)(prow + k0 + 16 + base);
#pragma unroll
      for (int e = 0; e < 8; ++e) a[e] = (bf16)r0[e];
#pragma unroll
      for (int e = 0; e < 8; ++e) a[8 + e] = (bf16)r1[e];
    }
#pragma unroll
    for (int t = 0; t < 8; ++t) {
      const int dim = wave * 128 + t * 16 + (lane & 15);
      ABPack bm;  // B column n = Vt row dim (contiguous over keys)
      const bf16* bp = Vt + ((size_t)(b * DB + dim)) * NSEQ + k0 + half * 8;
      bm.h[0] = *(const v8bf*)bp;
      bm.h[1] = *(const v8bf*)(bp + 16);
      oacc[t] = wmma_bf16(a, bm.v, oacc[t]);
    }
  }
#pragma unroll
  for (int t = 0; t < 8; ++t) {
    const int n = wave * 128 + t * 16 + (lane & 15);
#pragma unroll
    for (int r = 0; r < 8; ++r) {
      const int m = b * NSEQ + qrow0 + r + half * 8;
      O[(size_t)m * DB + n] = (bf16)oacc[t][r];
    }
  }
}

// ---------------------------------------------------------------------------
// Host orchestration
// ---------------------------------------------------------------------------
extern "C" void kernel_launch(void* const* d_in, const int* in_sizes, int n_in,
                              void* d_out, int out_size, void* d_ws, size_t ws_size,
                              hipStream_t stream) {
  const float* x_a   = (const float*)d_in[0];
  const float* x_b   = (const float*)d_in[1];
  const float* sa_wq = (const float*)d_in[2];  const float* sa_bq = (const float*)d_in[3];
  const float* sa_wk = (const float*)d_in[4];  const float* sa_bk = (const float*)d_in[5];
  const float* sa_wv = (const float*)d_in[6];  const float* sa_bv = (const float*)d_in[7];
  const float* sa_wo = (const float*)d_in[8];  const float* sa_bo = (const float*)d_in[9];
  const float* ca_wq = (const float*)d_in[10]; const float* ca_bq = (const float*)d_in[11];
  const float* ca_wk = (const float*)d_in[12]; const float* ca_bk = (const float*)d_in[13];
  const float* ca_wv = (const float*)d_in[14]; const float* ca_bv = (const float*)d_in[15];
  const float* ca_wo = (const float*)d_in[16]; const float* ca_bo = (const float*)d_in[17];

  char* ws = (char*)d_ws;
  size_t off = 0;
  auto alloc = [&](size_t bytes) -> void* {
    off = (off + 255) & ~(size_t)255;
    void* p = ws + off;
    off += bytes;
    return p;
  };

  bf16* Xa16   = (bf16*)alloc((size_t)MROWS * DA * 2);
  bf16* Xb16   = (bf16*)alloc((size_t)MROWS * DB * 2);
  bf16* WtSaQ  = (bf16*)alloc((size_t)DB * DB * 2);
  bf16* WtSaK  = (bf16*)alloc((size_t)DB * DB * 2);
  bf16* WtSaV  = (bf16*)alloc((size_t)DB * DB * 2);
  bf16* WtSaO  = (bf16*)alloc((size_t)DB * DB * 2);
  bf16* WtCaQ  = (bf16*)alloc((size_t)DB * DA * 2);   // [N=1024][K=768]
  bf16* WtCaK  = (bf16*)alloc((size_t)DB * DB * 2);
  bf16* WtCaV  = (bf16*)alloc((size_t)DB * DB * 2);
  bf16* WtCaO  = (bf16*)alloc((size_t)DB * DB * 2);
  bf16* Q16    = (bf16*)alloc((size_t)MROWS * DB * 2);
  bf16* K16    = (bf16*)alloc((size_t)MROWS * DB * 2);
  bf16* Vt16   = (bf16*)alloc((size_t)MROWS * DB * 2);
  bf16* Attn16 = (bf16*)alloc((size_t)MROWS * DB * 2);
  float* XbNew = (float*)alloc((size_t)MROWS * DB * 4);
  bf16* XbNew16= (bf16*)alloc((size_t)MROWS * DB * 2);

  const dim3 thr(256);
  cvt_f32_bf16_kernel<<<dim3((MROWS * DA + 255) / 256), thr, 0, stream>>>(x_a, Xa16, MROWS * DA);
  cvt_f32_bf16_kernel<<<dim3((MROWS * DB + 255) / 256), thr, 0, stream>>>(x_b, Xb16, MROWS * DB);
  transpose_cvt_kernel<<<dim3(DB / 16, DB / 16), thr, 0, stream>>>(sa_wq, WtSaQ, DB, DB);
  transpose_cvt_kernel<<<dim3(DB / 16, DB / 16), thr, 0, stream>>>(sa_wk, WtSaK, DB, DB);
  transpose_cvt_kernel<<<dim3(DB / 16, DB / 16), thr, 0, stream>>>(sa_wv, WtSaV, DB, DB);
  transpose_cvt_kernel<<<dim3(DB / 16, DB / 16), thr, 0, stream>>>(sa_wo, WtSaO, DB, DB);
  transpose_cvt_kernel<<<dim3(DB / 16, DA / 16), thr, 0, stream>>>(ca_wq, WtCaQ, DA, DB);
  transpose_cvt_kernel<<<dim3(DB / 16, DB / 16), thr, 0, stream>>>(ca_wk, WtCaK, DB, DB);
  transpose_cvt_kernel<<<dim3(DB / 16, DB / 16), thr, 0, stream>>>(ca_wv, WtCaV, DB, DB);
  transpose_cvt_kernel<<<dim3(DB / 16, DB / 16), thr, 0, stream>>>(ca_wo, WtCaO, DB, DB);

  const dim3 ggemm(MROWS / 32, DB / 256);
  const size_t lds_g1024 = (size_t)32 * DB * 2;   // 64KB A panel (K=1024)
  const size_t lds_g768  = (size_t)32 * DA * 2;   // 48KB A panel (K=768)
  const size_t attn_lds  = (size_t)16 * DB * 2 + (size_t)16 * NSEQ * 4;  // 160KB

  // ---- Self-attention on x_b ----
  gemm_bias_kernel<<<ggemm, thr, lds_g1024, stream>>>(Xb16, WtSaQ, sa_bq, Q16, nullptr, nullptr, nullptr, DB, DB, NSEQ);
  gemm_bias_kernel<<<ggemm, thr, lds_g1024, stream>>>(Xb16, WtSaK, sa_bk, K16, nullptr, nullptr, nullptr, DB, DB, NSEQ);
  gemm_bias_kernel<<<ggemm, thr, lds_g1024, stream>>>(Xb16, WtSaV, sa_bv, nullptr, nullptr, nullptr, Vt16, DB, DB, NSEQ);
  attn_kernel<<<dim3(BATCH * NSEQ / 16), thr, attn_lds, stream>>>(Q16, K16, Vt16, Attn16,
                                                                  1.0f / sqrtf((float)DB));
  gemm_bias_kernel<<<ggemm, thr, lds_g1024, stream>>>(Attn16, WtSaO, sa_bo, XbNew16, XbNew, x_b, nullptr, DB, DB, NSEQ);

  // ---- Cross-attention: q from x_a, k/v from x_b' ----
  gemm_bias_kernel<<<ggemm, thr, lds_g768, stream>>>(Xa16, WtCaQ, ca_bq, Q16, nullptr, nullptr, nullptr, DA, DB, NSEQ);
  gemm_bias_kernel<<<ggemm, thr, lds_g1024, stream>>>(XbNew16, WtCaK, ca_bk, K16, nullptr, nullptr, nullptr, DB, DB, NSEQ);
  gemm_bias_kernel<<<ggemm, thr, lds_g1024, stream>>>(XbNew16, WtCaV, ca_bv, nullptr, nullptr, nullptr, Vt16, DB, DB, NSEQ);
  attn_kernel<<<dim3(BATCH * NSEQ / 16), thr, attn_lds, stream>>>(Q16, K16, Vt16, Attn16,
                                                                  1.0f / sqrtf((float)DA));
  gemm_bias_kernel<<<ggemm, thr, lds_g1024, stream>>>(Attn16, WtCaO, ca_bo, nullptr, (float*)d_out, XbNew, nullptr, DB, DB, NSEQ);
}
